// GTLayer_28544352649804
// MI455X (gfx1250) — matrix operand order repaired and checked by reading, not compiled, not run on hardware
//
#include <hip/hip_runtime.h>
#include <hip/hip_bf16.h>

// ---------------------------------------------------------------------------
// Problem constants (fixed by the reference)
// ---------------------------------------------------------------------------
#define GN 40000      // nodes
#define GE 320000     // edges
#define GH 128        // hidden
#define NHEADS 4
#define DHEAD 32
#define TSTEPS 3

typedef __attribute__((ext_vector_type(16))) __bf16 v16bf;
typedef __attribute__((ext_vector_type(8)))  float  v8f;
typedef __attribute__((ext_vector_type(4)))  int    v4i;
typedef __attribute__((ext_vector_type(8)))  int    v8i;

union FragAB { v16bf v; uint4 q[2]; };
union Pack8  { unsigned short h[8]; uint4 q; };

__device__ __forceinline__ unsigned short f2bf(float f) {
    unsigned u = __float_as_uint(f);
    u += 0x7FFFu + ((u >> 16) & 1u);   // round-to-nearest-even
    return (unsigned short)(u >> 16);
}
__device__ __forceinline__ float bf2f(unsigned short h) {
    return __uint_as_float((unsigned)h << 16);
}
__device__ __forceinline__ unsigned enc_ord(float f) {
    unsigned u = __float_as_uint(f);
    return (u & 0x80000000u) ? ~u : (u | 0x80000000u);
}
__device__ __forceinline__ float dec_ord(unsigned o) {
    unsigned u = (o & 0x80000000u) ? (o & 0x7FFFFFFFu) : ~o;
    return __uint_as_float(u);
}
__device__ __forceinline__ float gelu_exact(float x) {
    return 0.5f * x * (1.0f + erff(x * 0.70710678118654752f));
}
__device__ __forceinline__ float sigmoidf(float x) {
    return 1.0f / (1.0f + __expf(-x));
}

// ---------------------------------------------------------------------------
// WMMA GEMM with TDM weight staging:
//   C[M,Nout] = act( A[M,K](f32) @ Wbf[Nout,K](bf16)^T + bias )
// A is converted f32->bf16 by VALU while staging to LDS; the bf16 weight tile
// (128 x 32) is DMA'd to LDS by the Tensor Data Mover (double buffered,
// issued by wave 0, tracked with TENSORcnt). TDM pad fields are set so the
// LDS row stride is exactly LDSW halves, matching the WMMA fragment layout.
// Block: 256 threads = 8 waves; block tile 64(M) x 128(N); wave tile 32x32.
// Requires: M%64==0, Nout%128==0, K%32==0 (true for all call sites).
// ---------------------------------------------------------------------------
#define LDSW 40   // padded row stride in halves (32 data + 8 pad)

__global__ __launch_bounds__(256)
void gemm_wmma_tdm(const float* __restrict__ A,
                   const unsigned short* __restrict__ Wbf,   // [Nout,K] bf16
                   const float* __restrict__ bias,           // may be nullptr
                   float* __restrict__ Cf,                   // f32 out (if Cbf==null)
                   unsigned short* __restrict__ Cbf,         // bf16 out (optional)
                   int M, int K, int Nout, int act)          // act: 0=none, 1=gelu
{
    __shared__ unsigned short Alds[64 * LDSW];
    __shared__ unsigned short Blds[2][128 * LDSW];

    const int tid  = threadIdx.x;
    const int lane = tid & 31;
    const int wave = tid >> 5;
    const int wm   = (wave & 1) * 32;       // wave row offset in block tile
    const int wn   = (wave >> 1) * 32;      // wave col offset in block tile
    const int row0 = blockIdx.y * 64;
    const int col0 = blockIdx.x * 128;

    const int g  = lane >> 4;               // lane-half select
    const int mr = lane & 15;               // A row / B col within 16-tile

    const int nsteps = K / 32;

    // ---- TDM descriptor issue (uniform scalar values -> SGPR tuples) ----
    auto tdm_issue = [&](int s, int b) {
        const unsigned short* gsrc = Wbf + (size_t)col0 * K + s * 32;
        unsigned long long ga = (unsigned long long)(uintptr_t)gsrc;
        unsigned lds = (unsigned)(uintptr_t)(void*)&Blds[b][0];
        v4i g0;
        g0[0] = 1;                                   // count=1, user descriptor
        g0[1] = (int)lds;                            // lds_addr
        g0[2] = (int)(unsigned)(ga & 0xFFFFFFFFull); // global_addr[31:0]
        g0[3] = (int)(((unsigned)(ga >> 32) & 0x01FFFFFFu) | (2u << 30)); // [56:32]|type=2
        v8i g1;
        // data_size=1(2B), pad_enable, pad_interval=3(16 DW = one 32-half row),
        // pad_amount=3(4 DW = 8 halves) -> LDS stride 40 halves
        g1[0] = (int)((1u << 16) | (1u << 20) | (3u << 22) | (3u << 25));
        const unsigned td0 = (unsigned)K;            // tensor_dim0 (elements)
        const unsigned td1 = 128u;                   // tensor_dim1 (rows)
        g1[1] = (int)((td0 & 0xFFFFu) << 16);                              // abar=0 | tdim0.lo
        g1[2] = (int)(((td0 >> 16) & 0xFFFFu) | ((td1 & 0xFFFFu) << 16));  // tdim0.hi | tdim1.lo
        g1[3] = (int)(((td1 >> 16) & 0xFFFFu) | (32u << 16));              // tdim1.hi | tile_dim0=32
        g1[4] = (int)(128u);                                               // tile_dim1=128, tile_dim2=0
        const unsigned long long st0 = (unsigned long long)K;              // dim0 stride (elements)
        g1[5] = (int)(unsigned)(st0 & 0xFFFFFFFFull);
        g1[6] = (int)(unsigned)((st0 >> 32) & 0xFFFFu);
        g1[7] = 0;
        asm volatile("tensor_load_to_lds %0, %1" :: "s"(g0), "s"(g1) : "memory");
    };

    if (wave == 0) tdm_issue(0, 0);

    v8f acc[2][2] = {};

    // A staging index map
    const int a_row = tid >> 2;             // 0..63
    const int a_col = (tid & 3) * 8;        // 0,8,16,24

    for (int s = 0; s < nsteps; ++s) {
        // ---- stage A tile (64 x 32), f32 -> bf16 in VALU ----
        {
            const float* src = A + (size_t)(row0 + a_row) * K + (s * 32 + a_col);
            float4 f0 = *(const float4*)(src);
            float4 f1 = *(const float4*)(src + 4);
            Pack8 p;
            p.h[0]=f2bf(f0.x); p.h[1]=f2bf(f0.y); p.h[2]=f2bf(f0.z); p.h[3]=f2bf(f0.w);
            p.h[4]=f2bf(f1.x); p.h[5]=f2bf(f1.y); p.h[6]=f2bf(f1.z); p.h[7]=f2bf(f1.w);
            *(uint4*)&Alds[a_row * LDSW + a_col] = p.q;
        }
        // ---- pipeline the weight DMA: issue next tile, wait for current ----
        if (wave == 0) {
            if (s + 1 < nsteps) {
                tdm_issue(s + 1, (s + 1) & 1);
                __builtin_amdgcn_s_wait_tensorcnt((short)1);  // tile s landed
            } else {
                __builtin_amdgcn_s_wait_tensorcnt((short)0);
            }
        }
        __syncthreads();

        const unsigned short* Bb = &Blds[s & 1][0];
        #pragma unroll
        for (int i = 0; i < 2; ++i) {
            FragAB fa;
            const int am = wm + i * 16 + mr;
            // A lane layout: k = [8g..8g+7] then [16+8g..16+8g+7]
            fa.q[0] = *(const uint4*)&Alds[am * LDSW + 8 * g];
            fa.q[1] = *(const uint4*)&Alds[am * LDSW + 8 * g + 16];
            #pragma unroll
            for (int j = 0; j < 2; ++j) {
                FragAB fb;
                const int bn = wn + j * 16 + mr;
                // B lane layout: k = [8g..8g+7] then [16+8g..16+8g+7]
                fb.q[0] = *(const uint4*)&Bb[bn * LDSW + 8 * g];
                fb.q[1] = *(const uint4*)&Bb[bn * LDSW + 8 * g + 16];
                acc[i][j] = __builtin_amdgcn_wmma_f32_16x16x32_bf16(
                    false, fa.v, false, fb.v, (short)0, acc[i][j], false, false);
            }
        }
        __syncthreads();
    }

    // ---- epilogue: bias (+gelu), store f32 or bf16 ----
    #pragma unroll
    for (int i = 0; i < 2; ++i) {
        #pragma unroll
        for (int j = 0; j < 2; ++j) {
            const int n = col0 + wn + j * 16 + mr;
            const float bv = bias ? bias[n] : 0.0f;
            #pragma unroll
            for (int r = 0; r < 8; ++r) {
                const int m = row0 + wm + i * 16 + r + 8 * g;
                float v = acc[i][j][r] + bv;
                if (act == 1) v = gelu_exact(v);
                if (Cbf) Cbf[(size_t)m * Nout + n] = f2bf(v);
                else     Cf [(size_t)m * Nout + n] = v;
            }
        }
    }
}

// ---------------------------------------------------------------------------
// One-time f32 -> bf16 conversion (weights)
// ---------------------------------------------------------------------------
__global__ __launch_bounds__(256)
void cvt_bf16_kernel(const float* __restrict__ in, unsigned short* __restrict__ out, int n)
{
    const int i = blockIdx.x * 256 + threadIdx.x;
    if (i < n) out[i] = f2bf(in[i]);
}

// ---------------------------------------------------------------------------
// Attention edge kernels (wave32 per edge)
// ---------------------------------------------------------------------------
__global__ __launch_bounds__(256)
void edge_alpha_kernel(const float* __restrict__ Qn, const float* __restrict__ Kn,
                       const unsigned short* __restrict__ EAk,
                       const int* __restrict__ src, const int* __restrict__ dst,
                       float* __restrict__ alpha, unsigned* __restrict__ segmax)
{
    const int e = blockIdx.x * 8 + (threadIdx.x >> 5);
    if (e >= GE) return;
    const int lane = threadIdx.x & 31;
    const int s = src[e], d = dst[e];

    float acc[NHEADS];
    #pragma unroll
    for (int h = 0; h < NHEADS; ++h) {
        const int f = h * DHEAD + lane;
        const float q = Qn[(size_t)d * GH + f];
        const float k = Kn[(size_t)s * GH + f] + bf2f(EAk[(size_t)e * GH + f]);
        acc[h] = q * k;
    }
    #pragma unroll
    for (int off = 16; off > 0; off >>= 1) {
        #pragma unroll
        for (int h = 0; h < NHEADS; ++h)
            acc[h] += __shfl_xor(acc[h], off, 32);
    }
    if (lane < NHEADS) {
        const float a = acc[lane] * 0.17677669529663687f;  // 1/sqrt(32)
        alpha[(size_t)e * NHEADS + lane] = a;
        atomicMax(&segmax[(size_t)d * NHEADS + lane], enc_ord(a));
    }
}

__global__ __launch_bounds__(256)
void edge_exp_kernel(const float* __restrict__ alpha,
                     const unsigned* __restrict__ segmax,
                     const int* __restrict__ dst,
                     float* __restrict__ aexp, float* __restrict__ segsum)
{
    const int idx = blockIdx.x * 256 + threadIdx.x;
    if (idx >= GE * NHEADS) return;
    const int e = idx >> 2, h = idx & 3;
    const int d = dst[e];
    const float m = dec_ord(segmax[(size_t)d * NHEADS + h]);
    const float a = __expf(alpha[idx] - m);
    aexp[idx] = a;
    atomicAdd(&segsum[(size_t)d * NHEADS + h], a);
}

__global__ __launch_bounds__(256)
void edge_scatter_kernel(const float* __restrict__ Vn,
                         const unsigned short* __restrict__ EAv,
                         const float* __restrict__ aexp,
                         const int* __restrict__ src, const int* __restrict__ dst,
                         float* __restrict__ agg)
{
    const int e = blockIdx.x * 8 + (threadIdx.x >> 5);
    if (e >= GE) return;
    const int lane = threadIdx.x & 31;
    const int s = src[e], d = dst[e];
    #pragma unroll
    for (int h = 0; h < NHEADS; ++h) {
        const float a = aexp[(size_t)e * NHEADS + h];
        const int f = h * DHEAD + lane;
        const float v = Vn[(size_t)s * GH + f] + bf2f(EAv[(size_t)e * GH + f]);
        atomicAdd(&agg[(size_t)d * GH + f], a * v);
    }
}

__global__ __launch_bounds__(256)
void normalize_kernel(float* __restrict__ agg, const float* __restrict__ segsum)
{
    const int idx = blockIdx.x * 256 + threadIdx.x;
    if (idx >= GN * GH) return;
    const int n = idx >> 7, f = idx & 127;
    agg[idx] = agg[idx] / (segsum[(size_t)n * NHEADS + (f >> 5)] + 1e-16f);
}

// ---------------------------------------------------------------------------
// LayerNorm over H=128: wave32 per row, 4 floats/lane, in-wave reductions.
// ---------------------------------------------------------------------------
__global__ __launch_bounds__(256)
void ln_kernel(const float* __restrict__ in, const float* __restrict__ res,
               const float* __restrict__ gw, const float* __restrict__ bw,
               float* __restrict__ out)
{
    const int row = blockIdx.x * 8 + (threadIdx.x >> 5);
    if (row >= GN) return;
    const int lane = threadIdx.x & 31;
    const int f = lane * 4;

    float4 v = *(const float4*)&in[(size_t)row * GH + f];
    if (res) {
        float4 r = *(const float4*)&res[(size_t)row * GH + f];
        v.x += r.x; v.y += r.y; v.z += r.z; v.w += r.w;
    }
    float s = v.x + v.y + v.z + v.w;
    #pragma unroll
    for (int off = 16; off > 0; off >>= 1) s += __shfl_xor(s, off, 32);
    const float mean = s * (1.0f / 128.0f);

    float4 c; c.x = v.x - mean; c.y = v.y - mean; c.z = v.z - mean; c.w = v.w - mean;
    float sq = c.x * c.x + c.y * c.y + c.z * c.z + c.w * c.w;
    #pragma unroll
    for (int off = 16; off > 0; off >>= 1) sq += __shfl_xor(sq, off, 32);
    const float rstd = rsqrtf(sq * (1.0f / 128.0f) + 1e-12f);

    float4 gv = *(const float4*)&gw[f];
    float4 bv = *(const float4*)&bw[f];
    float4 o;
    o.x = gv.x * c.x * rstd + bv.x;
    o.y = gv.y * c.y * rstd + bv.y;
    o.z = gv.z * c.z * rstd + bv.z;
    o.w = gv.w * c.w * rstd + bv.w;
    *(float4*)&out[(size_t)row * GH + f] = o;
}

// ---------------------------------------------------------------------------
// GRU gate fusion (torch order r,z,n): h = (1-z)*n + z*h
// ---------------------------------------------------------------------------
__global__ __launch_bounds__(256)
void gru_kernel(const float* __restrict__ gi, const float* __restrict__ gh,
                float* __restrict__ h)
{
    const int idx = blockIdx.x * 256 + threadIdx.x;
    if (idx >= GN * GH) return;
    const int n = idx >> 7, f = idx & 127;
    const size_t b = (size_t)n * (3 * GH);
    const float ir = gi[b + f],          hr = gh[b + f];
    const float iz = gi[b + GH + f],     hz = gh[b + GH + f];
    const float in_ = gi[b + 2*GH + f],  hn = gh[b + 2*GH + f];
    const float r = sigmoidf(ir + hr);
    const float z = sigmoidf(iz + hz);
    const float nn = tanhf(in_ + r * hn);
    h[idx] = (1.0f - z) * nn + z * h[idx];
}

// ---------------------------------------------------------------------------
// Host launcher
// ---------------------------------------------------------------------------
static inline void launch_gemm(const float* A, const unsigned short* Wbf,
                               const float* bias, float* Cf, unsigned short* Cbf,
                               int M, int K, int Nout, int act, hipStream_t stream)
{
    dim3 grid(Nout / 128, M / 64);
    gemm_wmma_tdm<<<grid, 256, 0, stream>>>(A, Wbf, bias, Cf, Cbf, M, K, Nout, act);
}

extern "C" void kernel_launch(void* const* d_in, const int* in_sizes, int n_in,
                              void* d_out, int out_size, void* d_ws, size_t ws_size,
                              hipStream_t stream)
{
    const float* x         = (const float*)d_in[0];
    const int*   eidx      = (const int*)  d_in[1];
    const float* edge_attr = (const float*)d_in[2];
    const float* wq = (const float*)d_in[3];  const float* bq = (const float*)d_in[4];
    const float* wk = (const float*)d_in[5];  const float* bk = (const float*)d_in[6];
    const float* wv = (const float*)d_in[7];  const float* bv = (const float*)d_in[8];
    const float* w_ao = (const float*)d_in[9];  const float* b_ao = (const float*)d_in[10];
    const float* ln1_g = (const float*)d_in[11]; const float* ln1_b = (const float*)d_in[12];
    const float* w_int = (const float*)d_in[13]; const float* b_int = (const float*)d_in[14];
    const float* w_out = (const float*)d_in[15]; const float* b_out = (const float*)d_in[16];
    const float* ln2_g = (const float*)d_in[17]; const float* ln2_b = (const float*)d_in[18];
    const float* w_ih = (const float*)d_in[19]; const float* w_hh = (const float*)d_in[20];
    const float* b_ih = (const float*)d_in[21]; const float* b_hh = (const float*)d_in[22];
    const float* ln3_g = (const float*)d_in[23]; const float* ln3_b = (const float*)d_in[24];

    const int* src = eidx;
    const int* dst = eidx + GE;
    float* out = (float*)d_out;

    // ---- carve scratch (256B-aligned slices) ----
    char* p = (char*)d_ws;
    auto allocf = [&p](size_t nfloats) -> float* {
        float* r = (float*)p;
        p += ((nfloats * sizeof(float) + 255) / 256) * 256;
        return r;
    };
    auto alloch = [&p](size_t nhalves) -> unsigned short* {
        unsigned short* r = (unsigned short*)p;
        p += ((nhalves * sizeof(unsigned short) + 255) / 256) * 256;
        return r;
    };

    unsigned short* EAk = alloch((size_t)GE * GH);   // bf16: halves edge traffic
    unsigned short* EAv = alloch((size_t)GE * GH);
    float* Qn    = allocf((size_t)GN * GH);
    float* Kn    = allocf((size_t)GN * GH);
    float* Vn    = allocf((size_t)GN * GH);
    float* x_cur = allocf((size_t)GN * GH);
    float* hbuf  = allocf((size_t)GN * GH);
    float* agg   = allocf((size_t)GN * GH);
    float* attp  = allocf((size_t)GN * GH);
    float* att   = allocf((size_t)GN * GH);
    float* ffo   = allocf((size_t)GN * GH);
    float* mbuf  = allocf((size_t)GN * GH);
    float* inter = allocf((size_t)GN * 4 * GH);
    float* gi    = allocf((size_t)GN * 3 * GH);
    float* ghb   = allocf((size_t)GN * 3 * GH);
    float* alpha = allocf((size_t)GE * NHEADS);
    float* aexp  = allocf((size_t)GE * NHEADS);
    unsigned* segmax = (unsigned*)allocf((size_t)GN * NHEADS);
    float* segsum = allocf((size_t)GN * NHEADS);
    // bf16 weight copies for the TDM B-path
    unsigned short* wq_b   = alloch(GH * GH);
    unsigned short* wk_b   = alloch(GH * GH);
    unsigned short* wv_b   = alloch(GH * GH);
    unsigned short* w_ao_b = alloch(GH * GH);
    unsigned short* w_int_b= alloch(4 * GH * GH);
    unsigned short* w_out_b= alloch(4 * GH * GH);
    unsigned short* w_ih_b = alloch(3 * GH * GH);
    unsigned short* w_hh_b = alloch(3 * GH * GH);

    const size_t nh_bytes = (size_t)GN * GH * sizeof(float);

    // ---- preamble: weight conversion, state init, loop-invariant projections
    auto cvt = [&](const float* w, unsigned short* wb, int n) {
        cvt_bf16_kernel<<<(n + 255) / 256, 256, 0, stream>>>(w, wb, n);
    };
    cvt(wq, wq_b, GH * GH);       cvt(wk, wk_b, GH * GH);
    cvt(wv, wv_b, GH * GH);       cvt(w_ao, w_ao_b, GH * GH);
    cvt(w_int, w_int_b, 4*GH*GH); cvt(w_out, w_out_b, 4*GH*GH);
    cvt(w_ih, w_ih_b, 3*GH*GH);   cvt(w_hh, w_hh_b, 3*GH*GH);

    hipMemcpyAsync(x_cur, x, nh_bytes, hipMemcpyDeviceToDevice, stream);
    hipMemcpyAsync(hbuf,  x, nh_bytes, hipMemcpyDeviceToDevice, stream);
    launch_gemm(edge_attr, wk_b, nullptr, nullptr, EAk, GE, GH, GH, 0, stream);
    launch_gemm(edge_attr, wv_b, nullptr, nullptr, EAv, GE, GH, GH, 0, stream);

    const int eb_edge = (GE + 7) / 8;
    const int eb_eh   = (GE * NHEADS + 255) / 256;
    const int nb_nh   = (GN * GH + 255) / 256;
    const int nb_row  = (GN + 7) / 8;

    for (int t = 0; t < TSTEPS; ++t) {
        // node-level Q/K/V projections
        launch_gemm(x_cur, wq_b, bq, Qn, nullptr, GN, GH, GH, 0, stream);
        launch_gemm(x_cur, wk_b, bk, Kn, nullptr, GN, GH, GH, 0, stream);
        launch_gemm(x_cur, wv_b, bv, Vn, nullptr, GN, GH, GH, 0, stream);

        // attention: segment softmax + aggregate
        hipMemsetAsync(segmax, 0, (size_t)GN * NHEADS * sizeof(unsigned), stream);
        hipMemsetAsync(segsum, 0, (size_t)GN * NHEADS * sizeof(float), stream);
        hipMemsetAsync(agg,    0, nh_bytes, stream);
        edge_alpha_kernel<<<eb_edge, 256, 0, stream>>>(Qn, Kn, EAk, src, dst, alpha, segmax);
        edge_exp_kernel<<<eb_eh, 256, 0, stream>>>(alpha, segmax, dst, aexp, segsum);
        edge_scatter_kernel<<<eb_edge, 256, 0, stream>>>(Vn, EAv, aexp, src, dst, agg);
        normalize_kernel<<<nb_nh, 256, 0, stream>>>(agg, segsum);

        // attention output projection + residual LN1
        launch_gemm(agg, w_ao_b, b_ao, attp, nullptr, GN, GH, GH, 0, stream);
        ln_kernel<<<nb_row, 256, 0, stream>>>(attp, x_cur, ln1_g, ln1_b, att);

        // FFN (gelu fused) + residual LN2
        launch_gemm(att, w_int_b, b_int, inter, nullptr, GN, GH, 4 * GH, 1, stream);
        launch_gemm(inter, w_out_b, b_out, ffo, nullptr, GN, 4 * GH, GH, 0, stream);
        ln_kernel<<<nb_row, 256, 0, stream>>>(ffo, att, ln2_g, ln2_b, mbuf);

        // GRU step + LN3
        launch_gemm(mbuf, w_ih_b, b_ih, gi,  nullptr, GN, GH, 3 * GH, 0, stream);
        launch_gemm(hbuf, w_hh_b, b_hh, ghb, nullptr, GN, GH, 3 * GH, 0, stream);
        gru_kernel<<<nb_nh, 256, 0, stream>>>(gi, ghb, hbuf);
        ln_kernel<<<nb_row, 256, 0, stream>>>(hbuf, nullptr, ln3_g, ln3_b,
                                              (t == TSTEPS - 1) ? out : x_cur);
    }
}
